// NativeAttentionModule_11020886082253
// MI455X (gfx1250) — matrix operand order repaired and checked by reading, not compiled, not run on hardware
//
#include <hip/hip_runtime.h>
#include <stdint.h>

typedef __bf16 bf16_t;
typedef __attribute__((ext_vector_type(16))) __bf16 v16bf;
typedef __attribute__((ext_vector_type(8)))  __bf16 v8bf;
typedef __attribute__((ext_vector_type(8)))  float  v8f;
typedef __attribute__((ext_vector_type(4)))  float  v4f;
typedef __attribute__((ext_vector_type(4)))  uint32_t u32x4;
typedef __attribute__((ext_vector_type(8)))  int      i32x8;
typedef __attribute__((ext_vector_type(4)))  int      i32x4;

#define S_LEN   2048
#define D_DIM   128
#define QTILE   64            // q rows per workgroup (4 waves x 16)
#define KVTILE  32            // kv rows per iteration
#define NWAVE   4
#define NTHREADS 128
#define NITER   (S_LEN / KVTILE)   // 64

// LDS layout (bytes)
#define KSTG_STRIDE 132                     // f32 per staged K row (TDM pads 4 DW per 128 DW)
#define KH_STRIDE   136                     // bf16 per Kh/Kl row (272B, 16B-divisible)
#define VT_STRIDE   40                      // bf16 per VT row (80B, 16B-divisible)

#define KSTG_OFF  0
#define KSTG_BYTES (KVTILE * KSTG_STRIDE * 4)          // 16896
#define KH_OFF    (KSTG_OFF + KSTG_BYTES)              // 16896
#define KH_BYTES  (KVTILE * KH_STRIDE * 2)             // 8704
#define KL_OFF    (KH_OFF + KH_BYTES)                  // 25600
#define VTH_OFF   (KL_OFF + KH_BYTES)                  // 34304
#define VT_BYTES  (D_DIM * VT_STRIDE * 2)              // 10240
#define VTL_OFF   (VTH_OFF + VT_BYTES)                 // 44544
#define P_OFF     (VTL_OFF + VT_BYTES)                 // 54784
#define P_BYTES_PER_WAVE 2048                          // 16x32 bf16 hi + lo
#define SMEM_BYTES (P_OFF + NWAVE * P_BYTES_PER_WAVE)  // 62976

__device__ __forceinline__ uint32_t lds_off32(const void* p) {
  return (uint32_t)(uintptr_t)p;   // low 32 bits of generic LDS pointer = LDS byte offset
}

// ---- Tensor Data Mover: one-shot DMA of a 32x128 f32 tile into LDS ----
// D# per CDNA5 ISA:
//   group0: count=1 | lds_addr | global_addr[56:0] | type=2
//   group1: data_size=4B, pad_enable (4 DWORDs pad per 128 DWORDs => 528B LDS rows),
//           tensor_dim0=128, tensor_dim1=32 (dims relative to tile start),
//           tile_dim0=128, tile_dim1=32, tensor_dim0_stride=128 elements
__device__ __forceinline__ void tdm_load_k_tile(const float* gaddr, uint32_t lds_addr) {
  uint64_t ga = (uint64_t)(uintptr_t)gaddr;
  u32x4 g0;
  g0[0] = 1u;                                   // count=1 (valid user descriptor)
  g0[1] = lds_addr;                             // lds_addr (bytes)
  g0[2] = (uint32_t)ga;                         // global_addr[31:0]
  g0[3] = (uint32_t)(ga >> 32) | (2u << 30);    // global_addr[56:32] | type=2
  i32x8 g1;
  g1[0] = (int)((2u << 16) | (1u << 20) | (6u << 22) | (3u << 25));
  //        data_size=4B | pad_enable | pad_interval=128DW | pad_amount=4DW
  g1[1] = (int)(128u << 16);                    // tensor_dim0 = 128 (bits 79:48 low half)
  g1[2] = (int)(32u << 16);                     // tensor_dim0 hi=0 | tensor_dim1 = 32
  g1[3] = (int)(128u << 16);                    // tensor_dim1 hi=0 | tile_dim0 = 128
  g1[4] = 32;                                   // tile_dim1 = 32 | tile_dim2 = 0
  g1[5] = 128;                                  // tensor_dim0_stride = 128 elements
  g1[6] = 0;
  g1[7] = 0;
  i32x4 gz4 = {0, 0, 0, 0};                     // groups 2/3 unused (2D tensor)
  i32x8 gz8 = {0, 0, 0, 0, 0, 0, 0, 0};
  __builtin_amdgcn_tensor_load_to_lds(g0, g1, gz4, gz4, gz8, 0);
}

// fp32 -> bf16 hi + bf16 lo (error-compensated split)
__device__ __forceinline__ bf16_t bf_hi(float x) { return (bf16_t)x; }
__device__ __forceinline__ bf16_t bf_lo(float x, bf16_t h) {
  return (bf16_t)(x - (float)h);
}

// concat two 16B-aligned v8bf LDS loads into a v16bf fragment (register concat)
__device__ __forceinline__ v16bf ld_frag(const bf16_t* p) {
  v8bf a = *(const v8bf*)p;
  v8bf b = *(const v8bf*)(p + 8);
  return __builtin_shufflevector(a, b, 0, 1, 2, 3, 4, 5, 6, 7,
                                 8, 9, 10, 11, 12, 13, 14, 15);
}

__global__ __launch_bounds__(NTHREADS)
void attn_relu2_wmma(const float* __restrict__ q, const float* __restrict__ kk,
                     const float* __restrict__ vv, const float* __restrict__ scale_p,
                     float* __restrict__ out) {
  __shared__ __align__(16) unsigned char smem[SMEM_BYTES];
  float*  Kstg = (float*)(smem + KSTG_OFF);    // [32][132] f32 staged K (TDM dest)
  bf16_t* Kh   = (bf16_t*)(smem + KH_OFF);     // [32][136] bf16, kv x d
  bf16_t* Kl   = (bf16_t*)(smem + KL_OFF);
  bf16_t* VTh  = (bf16_t*)(smem + VTH_OFF);    // [128][40] bf16, d x kv (V^T)
  bf16_t* VTl  = (bf16_t*)(smem + VTL_OFF);

  const int tid  = threadIdx.x;
  const int lane = tid & 31;
  const int wv   = tid >> 5;                   // 0..3
  const int mrow = lane & 15;
  const int hi16 = (lane >> 4) & 1;

  const int nqt = S_LEN / QTILE;               // 32
  const int qt  = blockIdx.x % nqt;
  const int bh  = blockIdx.x / nqt;

  const float* qb = q  + (size_t)bh * S_LEN * D_DIM;
  const float* kb = kk + (size_t)bh * S_LEN * D_DIM;
  const float* vb = vv + (size_t)bh * S_LEN * D_DIM;
  float*       ob = out + (size_t)bh * S_LEN * D_DIM;

  const float scale = *scale_p;
  const uint32_t kstg_lds = lds_off32(Kstg);

  bf16_t* Ph = (bf16_t*)(smem + P_OFF + wv * P_BYTES_PER_WAVE);
  bf16_t* Pl = Ph + 16 * 32;

  // convert-phase thread mapping
  const int krow  = tid >> 2;                  // 0..31 (K row)
  const int kcol0 = (tid & 3) * 32;            // 32 d per thread
  const int vrp   = tid >> 3;                  // 0..15 (V row pair 2rp,2rp+1)
  const int vc0   = (tid & 7) * 16;            // 16 d per thread

  // ---- Q tile (16 x 128 per wave) as bf16 hi/lo A-fragments, loaded once ----
  v16bf qh[4], ql[4];
  {
    const int qrow = qt * QTILE + wv * 16 + mrow;
    #pragma unroll
    for (int c = 0; c < 4; ++c) {
      const int dbase = c * 32 + hi16 * 16;
      const float* p0 = qb + (size_t)qrow * D_DIM + dbase;
      #pragma unroll
      for (int e = 0; e < 8; ++e) {
        float x0 = p0[e];
        float x1 = p0[e + 8];
        bf16_t h0 = bf_hi(x0);
        bf16_t h1 = bf_hi(x1);
        qh[c][e]     = h0;  ql[c][e]     = bf_lo(x0, h0);
        qh[c][e + 8] = h1;  ql[c][e + 8] = bf_lo(x1, h1);
      }
    }
  }

  v8f zero = {};
  v8f acc[8];
  #pragma unroll
  for (int t = 0; t < 8; ++t) acc[t] = zero;

  // V prefetch registers: rows {2*vrp, 2*vrp+1} x cols vc0..vc0+15
  v4f vreg[8];

  // ---- prologue: stage tile 0 ----
  if (wv == 0) tdm_load_k_tile(kb, kstg_lds);
  {
    const float* p0 = vb + (size_t)(2 * vrp) * D_DIM + vc0;
    #pragma unroll
    for (int i = 0; i < 4; ++i) {
      vreg[i]     = *(const v4f*)(p0 + 4 * i);
      vreg[4 + i] = *(const v4f*)(p0 + D_DIM + 4 * i);
    }
  }

  for (int T = 0; T < NITER; ++T) {
    __builtin_amdgcn_s_wait_tensorcnt(0);   // no-op for non-issuing waves
    __syncthreads();   // staging ready; all waves done computing previous tile

    // ---- convert K staging -> Kh/Kl ----
    {
      const float*  sp = Kstg + krow * KSTG_STRIDE + kcol0;
      bf16_t* dh = Kh + krow * KH_STRIDE + kcol0;
      bf16_t* dl = Kl + krow * KH_STRIDE + kcol0;
      #pragma unroll
      for (int i = 0; i < 32; ++i) {
        float x = sp[i];
        bf16_t h = bf_hi(x);
        dh[i] = h;
        dl[i] = bf_lo(x, h);
      }
    }
    // ---- convert V regs -> VTh/VTl (transposed) ----
    {
      #pragma unroll
      for (int i = 0; i < 16; ++i) {
        float x0 = vreg[i >> 2][i & 3];
        float x1 = vreg[4 + (i >> 2)][i & 3];
        bf16_t h0 = bf_hi(x0), h1 = bf_hi(x1);
        bf16_t l0 = bf_lo(x0, h0), l1 = bf_lo(x1, h1);
        const int d = vc0 + i;
        VTh[d * VT_STRIDE + 2 * vrp]     = h0;
        VTh[d * VT_STRIDE + 2 * vrp + 1] = h1;
        VTl[d * VT_STRIDE + 2 * vrp]     = l0;
        VTl[d * VT_STRIDE + 2 * vrp + 1] = l1;
      }
    }
    __syncthreads();   // bf16 tiles complete; staging fully consumed

    // ---- prefetch next tile via TDM + register V loads (overlap compute) ----
    if (T + 1 < NITER) {
      const int kvb = (T + 1) * KVTILE;
      if (wv == 0) tdm_load_k_tile(kb + (size_t)kvb * D_DIM, kstg_lds);
      const float* p0 = vb + (size_t)(kvb + 2 * vrp) * D_DIM + vc0;
      #pragma unroll
      for (int i = 0; i < 4; ++i) {
        vreg[i]     = *(const v4f*)(p0 + 4 * i);
        vreg[4 + i] = *(const v4f*)(p0 + D_DIM + 4 * i);
      }
    }

    // ---- S = Q K^T (two 16-col kv groups), relu^2, stash P hi/lo ----
    #pragma unroll
    for (int g = 0; g < 2; ++g) {
      v8f s = zero;
      #pragma unroll
      for (int c = 0; c < 4; ++c) {
        const int kvrow = g * 16 + mrow;
        const int dbase = c * 32 + hi16 * 16;
        v16bf bh_ = ld_frag(Kh + kvrow * KH_STRIDE + dbase);
        v16bf bl_ = ld_frag(Kl + kvrow * KH_STRIDE + dbase);
        s = __builtin_amdgcn_wmma_f32_16x16x32_bf16(false, qh[c], false, bh_, (short)0, s, false, false);
        s = __builtin_amdgcn_wmma_f32_16x16x32_bf16(false, qh[c], false, bl_, (short)0, s, false, false);
        s = __builtin_amdgcn_wmma_f32_16x16x32_bf16(false, ql[c], false, bh_, (short)0, s, false, false);
      }
      const int kvcol = g * 16 + mrow;
      #pragma unroll
      for (int r = 0; r < 8; ++r) {
        float t0 = fmaxf(s[r] * scale, 0.0f);
        t0 = t0 * t0;
        bf16_t h = bf_hi(t0);
        bf16_t l = bf_lo(t0, h);
        const int m = r + hi16 * 8;
        Ph[m * 32 + kvcol] = h;
        Pl[m * 32 + kvcol] = l;
      }
    }

    // ---- read P back in A-fragment layout (same-wave LDS ops in order) ----
    const int kvbase = hi16 * 16;
    v16bf pah, pal;
    {
      const bf16_t* p0 = Ph + mrow * 32 + kvbase;
      const bf16_t* p1 = Pl + mrow * 32 + kvbase;
      v8bf a0 = *(const v8bf*)p0;
      v8bf a1 = *(const v8bf*)(p0 + 8);
      v8bf b0 = *(const v8bf*)p1;
      v8bf b1 = *(const v8bf*)(p1 + 8);
      pah = __builtin_shufflevector(a0, a1, 0, 1, 2, 3, 4, 5, 6, 7,
                                    8, 9, 10, 11, 12, 13, 14, 15);
      pal = __builtin_shufflevector(b0, b1, 0, 1, 2, 3, 4, 5, 6, 7,
                                    8, 9, 10, 11, 12, 13, 14, 15);
    }

    // ---- O += P V over 8 d-tiles ----
    #pragma unroll
    for (int t = 0; t < 8; ++t) {
      const int dcol = t * 16 + mrow;
      const int koff = hi16 * 16;
      v16bf vh_ = ld_frag(VTh + dcol * VT_STRIDE + koff);
      v16bf vl_ = ld_frag(VTl + dcol * VT_STRIDE + koff);
      acc[t] = __builtin_amdgcn_wmma_f32_16x16x32_bf16(false, pah, false, vh_, (short)0, acc[t], false, false);
      acc[t] = __builtin_amdgcn_wmma_f32_16x16x32_bf16(false, pah, false, vl_, (short)0, acc[t], false, false);
      acc[t] = __builtin_amdgcn_wmma_f32_16x16x32_bf16(false, pal, false, vh_, (short)0, acc[t], false, false);
    }
  }

  // ---- write O: C/D layout -> row-major global ----
  #pragma unroll
  for (int t = 0; t < 8; ++t) {
    #pragma unroll
    for (int r = 0; r < 8; ++r) {
      const int m = r + hi16 * 8;
      const int qrow = qt * QTILE + wv * 16 + m;
      ob[(size_t)qrow * D_DIM + t * 16 + mrow] = acc[t][r];
    }
  }
}

extern "C" void kernel_launch(void* const* d_in, const int* in_sizes, int n_in,
                              void* d_out, int out_size, void* d_ws, size_t ws_size,
                              hipStream_t stream) {
  const float* q     = (const float*)d_in[0];
  const float* k     = (const float*)d_in[1];
  const float* v     = (const float*)d_in[2];
  const float* scale = (const float*)d_in[3];
  float* out = (float*)d_out;

  const int BH = 2 * 16;
  dim3 grid(BH * (S_LEN / QTILE));   // 32 * 32 = 1024 workgroups
  attn_relu2_wmma<<<grid, NTHREADS, 0, stream>>>(q, k, v, scale, out);
}
